// BlurAwareSwinAttentionBlock_35716948033642
// MI455X (gfx1250) — compile-verified, matrix operand order checked by reading, not run on hardware
//
#include <hip/hip_runtime.h>
#include <hip/hip_bf16.h>

typedef __bf16 bf16_t;
typedef __attribute__((ext_vector_type(16))) __bf16 bf16x16;
typedef __attribute__((ext_vector_type(8)))  float  f32x8;

union Frag16 { bf16x16 v; uint4 q[2]; bf16_t e[16]; };

__device__ __forceinline__ f32x8 zero8() {
  f32x8 z = {0.f,0.f,0.f,0.f,0.f,0.f,0.f,0.f};
  return z;
}

__device__ __forceinline__ f32x8 wmma_bf16(bf16x16 a, bf16x16 b, f32x8 c) {
  // D = A(16x32 bf16) x B(32x16 bf16) + C(16x16 f32)
  return __builtin_amdgcn_wmma_f32_16x16x32_bf16(false, a, false, b, (short)0, c,
                                                 false, false);
}

// A-fragment (row pattern) from LDS bf16 row-major buffer.
// lane<16: row = mTile*16+lane, K = {kb..kb+7, kb+16..kb+23}, kb = kChunk*32
// lane>=16: same row set, kb += 8  -> {8..15, 24..31}
__device__ __forceinline__ bf16x16 frag_row(const bf16_t* base, int strideE,
                                            int mTile, int kChunk, int lane) {
  int m  = mTile * 16 + (lane & 15);
  int kb = kChunk * 32 + ((lane & 16) ? 8 : 0);
  const bf16_t* p = base + m * strideE + kb;
  Frag16 f;
  f.q[0] = *reinterpret_cast<const uint4*>(p);        // k = kb .. kb+7
  f.q[1] = *reinterpret_cast<const uint4*>(p + 16);   // k = kb+16 .. kb+23
  return f.v;
}

// B-fragment (column pattern) gathered from LDS bf16 row-major buffer (K x N).
__device__ __forceinline__ bf16x16 frag_col(const bf16_t* base, int strideE,
                                            int kChunk, int nTile, int lane) {
  int n  = nTile * 16 + (lane & 15);
  int kb = kChunk * 32 + ((lane & 16) ? 8 : 0);
  Frag16 f;
#pragma unroll
  for (int i = 0; i < 16; ++i) {
    int k = kb + i + ((i & 8) ? 8 : 0);
    f.e[i] = base[k * strideE + n];
  }
  return f.v;
}

// B-fragment from pre-packed bf16 weights: frag = nTile*kt + kChunk,
// lane's 16 values contiguous -> two global b128 loads.
__device__ __forceinline__ bf16x16 frag_pk(const bf16_t* w, int frag, int lane) {
  const bf16_t* p = w + ((long)frag << 9) + (lane << 4);
  Frag16 f;
  f.q[0] = reinterpret_cast<const uint4*>(p)[0];
  f.q[1] = reinterpret_cast<const uint4*>(p)[1];
  return f.v;
}

// Store a 16x16 f32 accumulator tile into a row-major bf16 LDS buffer.
__device__ __forceinline__ void store_tile(bf16_t* base, int strideE, int mTile,
                                           int colBase, const f32x8& acc, int lane) {
  int m0 = mTile * 16 + ((lane & 16) ? 8 : 0);
  int c  = colBase + (lane & 15);
#pragma unroll
  for (int r = 0; r < 8; ++r) base[(m0 + r) * strideE + c] = (bf16_t)acc[r];
}

// ---------------------------------------------------------------------------
// Weight packer: f32 [K,N] row-major -> bf16 fragment-packed for frag_pk().
// ---------------------------------------------------------------------------
__global__ void pack_weights(const float* __restrict__ src, bf16_t* __restrict__ dst,
                             int K, int N) {
  int kt = K >> 5;
  long total = (long)K * N;
  for (long idx = (long)blockIdx.x * blockDim.x + threadIdx.x; idx < total;
       idx += (long)gridDim.x * blockDim.x) {
    int frag = (int)(idx >> 9);
    int rem  = (int)(idx & 511);
    int L = rem >> 4, i = rem & 15;
    int ntile = frag / kt;
    int kc    = frag - ntile * kt;
    int n  = L & 15;
    int hi = L >> 4;
    int kloc = hi * 8 + i + ((i & 8) ? 8 : 0);
    int k = (kc << 5) + kloc;
    dst[idx] = (bf16_t)src[(long)k * N + (ntile << 4) + n];
  }
}

// ---------------------------------------------------------------------------
// Fused per-window Swin block.  LDS layout (bytes):
// ---------------------------------------------------------------------------
#define SM_XRES 0        // f32 64x260 residual (xw, later x1)     66560 B
#define SM_XA   66560    // bf16 64x264 GEMM A (xw, later x1)      33792 B
#define SM_Q    100352   // bf16 64x264 q   (reused: attn-out)     33792 B
#define SM_K    134144   // bf16 64x264 k                          33792 B
#define SM_V    167936   // bf16 64x264 v                          33792 B
#define SM_P    201728   // bf16 8 waves x 64x72 softmax probs     73728 B
#define SM_HID  134144   // bf16 64x1032 FF hidden (aliases k/v/P) 132096 B
#define SM_BLUR 275456   // f32[64] (1 + blur)
#define SM_S1   275712   // f32[64] LN1 sum
#define SM_SQ1  275968   // f32[64] LN1 sumsq
#define SM_S2   276224   // f32[64] LN2 sum
#define SM_SQ2  276480   // f32[64] LN2 sumsq
#define SM_TOTAL 276736

__global__ __launch_bounds__(256) void swin_blur_block(
    const float* __restrict__ x, const float* __restrict__ blur_map,
    const bf16_t* __restrict__ wqkv, const float* __restrict__ bqkv,
    const bf16_t* __restrict__ wproj, const float* __restrict__ bproj,
    const bf16_t* __restrict__ wff1, const float* __restrict__ bff1,
    const bf16_t* __restrict__ wff2, const float* __restrict__ bff2,
    const float* __restrict__ n1g, const float* __restrict__ n1b,
    const float* __restrict__ n2g, const float* __restrict__ n2b,
    float* __restrict__ out) {
  extern __shared__ char smem[];
  float*  xres = reinterpret_cast<float*>(smem + SM_XRES);
  bf16_t* xa   = reinterpret_cast<bf16_t*>(smem + SM_XA);
  bf16_t* qb   = reinterpret_cast<bf16_t*>(smem + SM_Q);
  bf16_t* kbuf = reinterpret_cast<bf16_t*>(smem + SM_K);
  bf16_t* vbuf = reinterpret_cast<bf16_t*>(smem + SM_V);
  bf16_t* pbuf = reinterpret_cast<bf16_t*>(smem + SM_P);
  bf16_t* hid  = reinterpret_cast<bf16_t*>(smem + SM_HID);
  bf16_t* ao   = qb;  // attention output aliases q (after barrier)
  float* blurf = reinterpret_cast<float*>(smem + SM_BLUR);
  float* s1    = reinterpret_cast<float*>(smem + SM_S1);
  float* sq1   = reinterpret_cast<float*>(smem + SM_SQ1);
  float* s2    = reinterpret_cast<float*>(smem + SM_S2);
  float* sq2   = reinterpret_cast<float*>(smem + SM_SQ2);

  const int tid  = threadIdx.x;
  const int lane = tid & 31;
  const int wave = tid >> 5;
  const int ln   = lane & 15;
  const int hi8  = (lane & 16) ? 8 : 0;
  const int wid  = blockIdx.x;
  const int bI   = wid >> 8;
  const int wy   = (wid >> 4) & 15;
  const int wx   = wid & 15;

  // ---------------- stage 0: load window, blur, zero LN stats --------------
  {
    const float* xb = x + (long)bI * 256 * 128 * 128;
    for (int it = 0; it < 64; ++it) {
      int idx = (it << 8) + tid;        // (c, t)
      int c = idx >> 6, t = idx & 63;
      int yy = wy * 8 + (t >> 3), xx = wx * 8 + (t & 7);
      float v = xb[((long)c * 128 + yy) * 128 + xx];
      xres[t * 260 + c] = v;
      xa[t * 264 + c]   = (bf16_t)v;
    }
    if (tid < 64) {
      int t = tid;
      int yy = wy * 8 + (t >> 3), xx = wx * 8 + (t & 7);
      float fy = fminf(fmaxf(yy * 0.25f - 0.375f, 0.f), 31.f);
      float fx = fminf(fmaxf(xx * 0.25f - 0.375f, 0.f), 31.f);
      int y0 = (int)fy, x0 = (int)fx;
      float ay = fy - y0, axf = fx - x0;
      int y1 = min(y0 + 1, 31), x1 = min(x0 + 1, 31);
      const float* bm = blur_map + bI * 1024;
      float bv = (1.f - ay) * ((1.f - axf) * bm[y0 * 32 + x0] + axf * bm[y0 * 32 + x1])
               + ay * ((1.f - axf) * bm[y1 * 32 + x0] + axf * bm[y1 * 32 + x1]);
      blurf[t] = 1.f + bv;      // BLUR_STRENGTH = 1
      s1[t] = 0.f; sq1[t] = 0.f; s2[t] = 0.f; sq2[t] = 0.f;
    }
  }
  __syncthreads();

  // ---------------- stage 1: QKV GEMM  (64x256) @ (256x768) ----------------
  for (int pair = 0; pair < 3; ++pair) {
    int ct0 = wave * 6 + pair * 2;
    f32x8 acc[4][2];
#pragma unroll
    for (int mt = 0; mt < 4; ++mt)
#pragma unroll
      for (int j = 0; j < 2; ++j) acc[mt][j] = zero8();
    for (int kc = 0; kc < 8; ++kc) {
      bf16x16 a[4];
#pragma unroll
      for (int mt = 0; mt < 4; ++mt) a[mt] = frag_row(xa, 264, mt, kc, lane);
      __builtin_prefetch(wqkv + (((long)ct0 * 8 + kc + 2) << 9), 0, 1);
#pragma unroll
      for (int j = 0; j < 2; ++j) {
        bf16x16 bw = frag_pk(wqkv, (ct0 + j) * 8 + kc, lane);
#pragma unroll
        for (int mt = 0; mt < 4; ++mt) acc[mt][j] = wmma_bf16(a[mt], bw, acc[mt][j]);
      }
    }
#pragma unroll
    for (int j = 0; j < 2; ++j) {
      int ct = ct0 + j;
      float bias = bqkv[ct * 16 + ln];
      bf16_t* dst = (ct < 16) ? qb : ((ct < 32) ? kbuf : vbuf);
      int col = (ct & 15) * 16;
#pragma unroll
      for (int mt = 0; mt < 4; ++mt) {
        f32x8 t = acc[mt][j];
#pragma unroll
        for (int r = 0; r < 8; ++r) t[r] += bias;
        store_tile(dst, 264, mt, col, t, lane);
      }
    }
  }
  __syncthreads();

  // ---------------- stage 2: attention (one head per wave) -----------------
  {
    int h = wave;  // head; its hd=32 slice is exactly K-chunk h
    f32x8 at[4][4];
    bf16x16 ak[4];
#pragma unroll
    for (int mt = 0; mt < 4; ++mt) ak[mt] = frag_row(kbuf, 264, mt, h, lane);
#pragma unroll
    for (int nt = 0; nt < 4; ++nt) {
      bf16x16 bq = frag_row(qb, 264, nt, h, lane);  // B = q^T (lane = query)
#pragma unroll
      for (int mt = 0; mt < 4; ++mt) at[mt][nt] = wmma_bf16(ak[mt], bq, zero8());
    }
    __syncthreads();  // q/k now dead; attn-out may alias q

    // scale (C^-0.5 = 1/16) and blur modulation along key axis (= rows here)
#pragma unroll
    for (int mt = 0; mt < 4; ++mt)
#pragma unroll
      for (int r = 0; r < 8; ++r) {
        float f = 0.0625f * blurf[mt * 16 + hi8 + r];
#pragma unroll
        for (int nt = 0; nt < 4; ++nt) at[mt][nt][r] *= f;
      }

    // softmax over keys (registers + one xor-16 shuffle), write P^T to LDS
    bf16_t* pw = pbuf + wave * (64 * 72);
#pragma unroll
    for (int nt = 0; nt < 4; ++nt) {
      float mx = -3.0e38f;
#pragma unroll
      for (int mt = 0; mt < 4; ++mt)
#pragma unroll
        for (int r = 0; r < 8; ++r) mx = fmaxf(mx, at[mt][nt][r]);
      mx = fmaxf(mx, __shfl_xor(mx, 16, 32));
      float sum = 0.f;
#pragma unroll
      for (int mt = 0; mt < 4; ++mt)
#pragma unroll
        for (int r = 0; r < 8; ++r) {
          float e = __expf(at[mt][nt][r] - mx);
          at[mt][nt][r] = e;
          sum += e;
        }
      sum += __shfl_xor(sum, 16, 32);
      float inv = 1.f / sum;
      int q = nt * 16 + ln;
#pragma unroll
      for (int mt = 0; mt < 4; ++mt) {
        int m0 = mt * 16 + hi8;
#pragma unroll
        for (int r = 0; r < 8; ++r)
          pw[q * 72 + m0 + r] = (bf16_t)(at[mt][nt][r] * inv);  // [query][key]
      }
    }

    // out = P^T @ v   (64x64) @ (64x32)
    f32x8 o[4][2];
#pragma unroll
    for (int mt = 0; mt < 4; ++mt)
#pragma unroll
      for (int j = 0; j < 2; ++j) o[mt][j] = zero8();
#pragma unroll
    for (int kc = 0; kc < 2; ++kc) {
      bf16x16 pa[4];
#pragma unroll
      for (int mt = 0; mt < 4; ++mt) pa[mt] = frag_row(pw, 72, mt, kc, lane);
#pragma unroll
      for (int j = 0; j < 2; ++j) {
        bf16x16 bv = frag_col(vbuf, 264, kc, h * 2 + j, lane);
#pragma unroll
        for (int mt = 0; mt < 4; ++mt) o[mt][j] = wmma_bf16(pa[mt], bv, o[mt][j]);
      }
    }
#pragma unroll
    for (int j = 0; j < 2; ++j)
#pragma unroll
      for (int mt = 0; mt < 4; ++mt)
        store_tile(ao, 264, mt, h * 32 + j * 16, o[mt][j], lane);
  }
  __syncthreads();

  // ---------------- stage 3: proj + LN1 + residual -------------------------
  {
    f32x8 p[4][2];
#pragma unroll
    for (int mt = 0; mt < 4; ++mt)
#pragma unroll
      for (int j = 0; j < 2; ++j) p[mt][j] = zero8();
    for (int kc = 0; kc < 8; ++kc) {
      bf16x16 a[4];
#pragma unroll
      for (int mt = 0; mt < 4; ++mt) a[mt] = frag_row(ao, 264, mt, kc, lane);
#pragma unroll
      for (int j = 0; j < 2; ++j) {
        bf16x16 bw = frag_pk(wproj, (wave * 2 + j) * 8 + kc, lane);
#pragma unroll
        for (int mt = 0; mt < 4; ++mt) p[mt][j] = wmma_bf16(a[mt], bw, p[mt][j]);
      }
    }
#pragma unroll
    for (int j = 0; j < 2; ++j) {
      float bias = bproj[(wave * 2 + j) * 16 + ln];
#pragma unroll
      for (int mt = 0; mt < 4; ++mt)
#pragma unroll
        for (int r = 0; r < 8; ++r) p[mt][j][r] += bias;
    }
    // LN1 partial sums (per token row) -> LDS float atomics
#pragma unroll
    for (int mt = 0; mt < 4; ++mt)
#pragma unroll
      for (int r = 0; r < 8; ++r) {
        float v0 = p[mt][0][r], v1 = p[mt][1][r];
        float s = v0 + v1, s2v = v0 * v0 + v1 * v1;
#pragma unroll
        for (int off = 1; off < 16; off <<= 1) {
          s   += __shfl_xor(s, off, 32);
          s2v += __shfl_xor(s2v, off, 32);
        }
        if (ln == 0) {
          int row = mt * 16 + hi8 + r;
          atomicAdd(&s1[row], s);
          atomicAdd(&sq1[row], s2v);
        }
      }
    __syncthreads();
#pragma unroll
    for (int j = 0; j < 2; ++j) {
      int cb = (wave * 2 + j) * 16 + ln;
      float g = n1g[cb], bb = n1b[cb];
#pragma unroll
      for (int mt = 0; mt < 4; ++mt)
#pragma unroll
        for (int r = 0; r < 8; ++r) {
          int row = mt * 16 + hi8 + r;
          float mean = s1[row] * (1.f / 256.f);
          float var  = sq1[row] * (1.f / 256.f) - mean * mean;
          float x1v  = xres[row * 260 + cb] +
                       (p[mt][j][r] - mean) * rsqrtf(var + 1e-5f) * g + bb;
          xres[row * 260 + cb] = x1v;       // residual for stage 5
          xa[row * 264 + cb]   = (bf16_t)x1v;  // A for FF1
        }
    }
  }
  __syncthreads();

  // ---------------- stage 4: FF1 + ReLU  (64x256)@(256x1024) ---------------
  for (int pair = 0; pair < 4; ++pair) {
    int ct0 = wave * 8 + pair * 2;
    f32x8 acc[4][2];
#pragma unroll
    for (int mt = 0; mt < 4; ++mt)
#pragma unroll
      for (int j = 0; j < 2; ++j) acc[mt][j] = zero8();
    for (int kc = 0; kc < 8; ++kc) {
      bf16x16 a[4];
#pragma unroll
      for (int mt = 0; mt < 4; ++mt) a[mt] = frag_row(xa, 264, mt, kc, lane);
#pragma unroll
      for (int j = 0; j < 2; ++j) {
        bf16x16 bw = frag_pk(wff1, (ct0 + j) * 8 + kc, lane);
#pragma unroll
        for (int mt = 0; mt < 4; ++mt) acc[mt][j] = wmma_bf16(a[mt], bw, acc[mt][j]);
      }
    }
#pragma unroll
    for (int j = 0; j < 2; ++j) {
      int ct = ct0 + j;
      float bias = bff1[ct * 16 + ln];
#pragma unroll
      for (int mt = 0; mt < 4; ++mt) {
        int m0 = mt * 16 + hi8;
#pragma unroll
        for (int r = 0; r < 8; ++r) {
          float v = fmaxf(acc[mt][j][r] + bias, 0.f);
          hid[(m0 + r) * 1032 + ct * 16 + ln] = (bf16_t)v;
        }
      }
    }
  }
  __syncthreads();

  // ---------------- stage 5: FF2 + LN2 + residual + store ------------------
  {
    f32x8 acc[4][2];
#pragma unroll
    for (int mt = 0; mt < 4; ++mt)
#pragma unroll
      for (int j = 0; j < 2; ++j) acc[mt][j] = zero8();
    for (int kc = 0; kc < 32; ++kc) {
      bf16x16 a[4];
#pragma unroll
      for (int mt = 0; mt < 4; ++mt) a[mt] = frag_row(hid, 1032, mt, kc, lane);
#pragma unroll
      for (int j = 0; j < 2; ++j) {
        bf16x16 bw = frag_pk(wff2, (wave * 2 + j) * 32 + kc, lane);
#pragma unroll
        for (int mt = 0; mt < 4; ++mt) acc[mt][j] = wmma_bf16(a[mt], bw, acc[mt][j]);
      }
    }
#pragma unroll
    for (int j = 0; j < 2; ++j) {
      float bias = bff2[(wave * 2 + j) * 16 + ln];
#pragma unroll
      for (int mt = 0; mt < 4; ++mt)
#pragma unroll
        for (int r = 0; r < 8; ++r) acc[mt][j][r] += bias;
    }
#pragma unroll
    for (int mt = 0; mt < 4; ++mt)
#pragma unroll
      for (int r = 0; r < 8; ++r) {
        float v0 = acc[mt][0][r], v1 = acc[mt][1][r];
        float s = v0 + v1, s2v = v0 * v0 + v1 * v1;
#pragma unroll
        for (int off = 1; off < 16; off <<= 1) {
          s   += __shfl_xor(s, off, 32);
          s2v += __shfl_xor(s2v, off, 32);
        }
        if (ln == 0) {
          int row = mt * 16 + hi8 + r;
          atomicAdd(&s2[row], s);
          atomicAdd(&sq2[row], s2v);
        }
      }
    __syncthreads();
#pragma unroll
    for (int j = 0; j < 2; ++j) {
      int cb = (wave * 2 + j) * 16 + ln;
      float g = n2g[cb], bb = n2b[cb];
#pragma unroll
      for (int mt = 0; mt < 4; ++mt)
#pragma unroll
        for (int r = 0; r < 8; ++r) {
          int row = mt * 16 + hi8 + r;
          float mean = s2[row] * (1.f / 256.f);
          float var  = sq2[row] * (1.f / 256.f) - mean * mean;
          float x2 = xres[row * 260 + cb] +
                     (acc[mt][j][r] - mean) * rsqrtf(var + 1e-5f) * g + bb;
          int yy = wy * 8 + (row >> 3), xx = wx * 8 + (row & 7);
          out[(((long)bI * 256 + cb) * 128 + yy) * 128 + xx] = x2;
        }
    }
  }
}

// ---------------------------------------------------------------------------
extern "C" void kernel_launch(void* const* d_in, const int* in_sizes, int n_in,
                              void* d_out, int out_size, void* d_ws, size_t ws_size,
                              hipStream_t stream) {
  const float* x     = (const float*)d_in[0];
  const float* blur  = (const float*)d_in[1];
  const float* qkvw  = (const float*)d_in[2];
  const float* qkvb  = (const float*)d_in[3];
  const float* projw = (const float*)d_in[4];
  const float* projb = (const float*)d_in[5];
  const float* ff1w  = (const float*)d_in[6];
  const float* ff1b  = (const float*)d_in[7];
  const float* ff2w  = (const float*)d_in[8];
  const float* ff2b  = (const float*)d_in[9];
  const float* n1g   = (const float*)d_in[10];
  const float* n1b   = (const float*)d_in[11];
  const float* n2g   = (const float*)d_in[12];
  const float* n2b   = (const float*)d_in[13];

  bf16_t* ws    = (bf16_t*)d_ws;
  bf16_t* pqkv  = ws;                 // 256x768  = 196608 elems
  bf16_t* pproj = ws + 196608;        // 256x256  =  65536
  bf16_t* pff1  = ws + 262144;        // 256x1024 = 262144
  bf16_t* pff2  = ws + 524288;        // 1024x256 = 262144  (total 1.5 MB)

  pack_weights<<<768,  256, 0, stream>>>(qkvw,  pqkv,  256, 768);
  pack_weights<<<256,  256, 0, stream>>>(projw, pproj, 256, 256);
  pack_weights<<<1024, 256, 0, stream>>>(ff1w,  pff1,  256, 1024);
  pack_weights<<<1024, 256, 0, stream>>>(ff2w,  pff2,  1024, 256);

  swin_blur_block<<<2048, 256, SM_TOTAL, stream>>>(
      x, blur, pqkv, qkvb, pproj, projb, pff1, ff1b, pff2, ff2b,
      n1g, n1b, n2g, n2b, (float*)d_out);
}
